// MultiHeadAttention_29472065585543
// MI455X (gfx1250) — matrix operand order repaired and checked by reading, not compiled, not run on hardware
//
#include <hip/hip_runtime.h>
#include <hip/hip_bf16.h>
#include <math.h>

// Problem constants (match reference)
#define Bv 4
#define Sv 2048
#define Dv 768
#define Hv 12
#define HDv 64

typedef __attribute__((ext_vector_type(16))) _Float16 v16h;
typedef __attribute__((ext_vector_type(8)))  _Float16 v8h;
typedef __attribute__((ext_vector_type(8)))  float    v8f;

// ---------------------------------------------------------------------------
// WMMA helpers (gfx1250, wave32). Layouts per CDNA5 ISA §7.12.2:
//  A (16x32 f16): lane L holds row M=L&15; half=L>>4.
//    v16h element i: i<8  -> K = k0 + half*8 + i
//                    i>=8 -> K = k0 + 16 + half*8 + (i-8)
//  B (32x16 f16): lane L holds col N=L&15; half=L>>4.
//    v16h element i: same K mapping as A (i<8 -> half*8+i, i>=8 -> 16+half*8+(i-8))
//  C/D (16x16 f32): reg r, lane L -> (M = r + 8*half, N = L&15)
// ---------------------------------------------------------------------------

static __device__ __forceinline__ v8f wmma_f16(v16h a, v16h b, v8f c) {
    return __builtin_amdgcn_wmma_f32_16x16x32_f16(
        /*neg_a=*/false, a, /*neg_b=*/false, b,
        /*c_mod=*/(short)0, c, /*reuse_a=*/false, /*reuse_b=*/false);
}

// base points at element [row0, k0] of a row-major (rows x K) f16 matrix, stride ld
static __device__ __forceinline__ v16h load_a_frag(const _Float16* base, int ld, int lane) {
    const int row  = lane & 15;
    const int half = lane >> 4;
    const _Float16* p = base + row * ld + half * 8;
    v8h lo = *(const v8h*)(p);
    v8h hi = *(const v8h*)(p + 16);
    v16h a;
#pragma unroll
    for (int i = 0; i < 8; ++i) { a[i] = lo[i]; a[i + 8] = hi[i]; }
    return a;
}

// B fragment: element (k,n) read from memory where the N-dimension strides by ld
// and K is contiguous. base points at [n0, k0].
static __device__ __forceinline__ v16h load_b_frag(const _Float16* base, int ld, int lane) {
    const int n    = lane & 15;
    const int half = lane >> 4;
    const _Float16* p = base + n * ld + half * 8;
    v8h lo = *(const v8h*)(p);
    v8h hi = *(const v8h*)(p + 16);
    v16h b;
#pragma unroll
    for (int i = 0; i < 8; ++i) { b[i] = lo[i]; b[i + 8] = hi[i]; }
    return b;
}

// ---------------------------------------------------------------------------
// Kernel 0: fp32 -> f16 conversion for x and the four weight matrices
// ---------------------------------------------------------------------------
__global__ void convert_f16_kernel(const float* __restrict__ x,
                                   const float* __restrict__ wq, const float* __restrict__ wk,
                                   const float* __restrict__ wv, const float* __restrict__ wo,
                                   _Float16* __restrict__ xh,
                                   _Float16* __restrict__ wqh, _Float16* __restrict__ wkh,
                                   _Float16* __restrict__ wvh, _Float16* __restrict__ woh) {
    const size_t NX = (size_t)Bv * Sv * Dv;
    const size_t NW = (size_t)Dv * Dv;
    size_t i = (size_t)blockIdx.x * blockDim.x + threadIdx.x;
    if (i < NX) { xh[i] = (_Float16)x[i]; return; }
    size_t j = i - NX;
    if (j < NW) { wqh[j] = (_Float16)wq[j]; return; }
    j -= NW;
    if (j < NW) { wkh[j] = (_Float16)wk[j]; return; }
    j -= NW;
    if (j < NW) { wvh[j] = (_Float16)wv[j]; return; }
    j -= NW;
    if (j < NW) { woh[j] = (_Float16)wo[j]; return; }
}

// ---------------------------------------------------------------------------
// Kernel 1: fused QKV projection. grid = (BS/32, 3, 3), block = 128 (4 waves).
// Each wave: 32x64 output tile, software-pipelined K-loop over D in steps
// of 32 (8 WMMAs per step, next step's fragments prefetched before use).
// Q,K stored (B,H,S,HD) f16; V stored transposed (B,H,HD,S) f16.
// ---------------------------------------------------------------------------
__global__ __launch_bounds__(128) void qkv_proj_kernel(
    const _Float16* __restrict__ xh,
    const _Float16* __restrict__ wqh, const _Float16* __restrict__ wkh,
    const _Float16* __restrict__ wvh,
    const float* __restrict__ bq, const float* __restrict__ bk, const float* __restrict__ bv,
    _Float16* __restrict__ Qh, _Float16* __restrict__ Kh, _Float16* __restrict__ Vt) {

    const int lane = threadIdx.x & 31;
    const int wave = threadIdx.x >> 5;
    const int m0 = blockIdx.x * 32;
    const int n0 = blockIdx.y * 256 + wave * 64;
    const int z  = blockIdx.z;

    const _Float16* wh = (z == 0) ? wqh : ((z == 1) ? wkh : wvh);
    const float* bias  = (z == 0) ? bq  : ((z == 1) ? bk  : bv);

    const _Float16* abase = xh + (size_t)m0 * Dv;
    const _Float16* bbase = wh + (size_t)n0 * Dv;

    v8f acc[8] = {v8f{}, v8f{}, v8f{}, v8f{}, v8f{}, v8f{}, v8f{}, v8f{}};

    // --- software pipeline: preload k=0 fragments ---
    v16h a0 = load_a_frag(abase, Dv, lane);
    v16h a1 = load_a_frag(abase + (size_t)16 * Dv, Dv, lane);
    v16h b0 = load_b_frag(bbase + (size_t)0  * Dv, Dv, lane);
    v16h b1 = load_b_frag(bbase + (size_t)16 * Dv, Dv, lane);
    v16h b2 = load_b_frag(bbase + (size_t)32 * Dv, Dv, lane);
    v16h b3 = load_b_frag(bbase + (size_t)48 * Dv, Dv, lane);

    for (int k0 = 0; k0 < Dv; k0 += 32) {
        // prefetch next step (clamped: last iteration redundantly reloads)
        const int kn = (k0 + 32 < Dv) ? (k0 + 32) : k0;
        v16h na0 = load_a_frag(abase + kn, Dv, lane);
        v16h na1 = load_a_frag(abase + (size_t)16 * Dv + kn, Dv, lane);
        v16h nb0 = load_b_frag(bbase + (size_t)0  * Dv + kn, Dv, lane);
        v16h nb1 = load_b_frag(bbase + (size_t)16 * Dv + kn, Dv, lane);
        v16h nb2 = load_b_frag(bbase + (size_t)32 * Dv + kn, Dv, lane);
        v16h nb3 = load_b_frag(bbase + (size_t)48 * Dv + kn, Dv, lane);

        acc[0] = wmma_f16(a0, b0, acc[0]);
        acc[1] = wmma_f16(a0, b1, acc[1]);
        acc[2] = wmma_f16(a0, b2, acc[2]);
        acc[3] = wmma_f16(a0, b3, acc[3]);
        acc[4] = wmma_f16(a1, b0, acc[4]);
        acc[5] = wmma_f16(a1, b1, acc[5]);
        acc[6] = wmma_f16(a1, b2, acc[6]);
        acc[7] = wmma_f16(a1, b3, acc[7]);

        a0 = na0; a1 = na1; b0 = nb0; b1 = nb1; b2 = nb2; b3 = nb3;
    }

    const int half = lane >> 4;
    const int cl   = lane & 15;
    const int bb   = m0 / Sv;       // 32 divides S, so all rows share the batch
    const int sb   = m0 % Sv;
#pragma unroll
    for (int f = 0; f < 4; ++f) {
        const int n  = n0 + f * 16 + cl;
        const float bval = bias[n];
        const int h = n / HDv;
        const int d = n % HDv;
#pragma unroll
        for (int r = 0; r < 8; ++r) {
            const int s0r = sb + r + 8 * half;        // rows m0..m0+15
            const int s1r = s0r + 16;                 // rows m0+16..m0+31
            const _Float16 hv0 = (_Float16)(acc[f][r] + bval);
            const _Float16 hv1 = (_Float16)(acc[4 + f][r] + bval);
            if (z == 0) {
                Qh[(((size_t)bb * Hv + h) * Sv + s0r) * HDv + d] = hv0;
                Qh[(((size_t)bb * Hv + h) * Sv + s1r) * HDv + d] = hv1;
            } else if (z == 1) {
                Kh[(((size_t)bb * Hv + h) * Sv + s0r) * HDv + d] = hv0;
                Kh[(((size_t)bb * Hv + h) * Sv + s1r) * HDv + d] = hv1;
            } else {
                Vt[(((size_t)bb * Hv + h) * HDv + d) * Sv + s0r] = hv0;
                Vt[(((size_t)bb * Hv + h) * HDv + d) * Sv + s1r] = hv1;
            }
        }
    }
}

// ---------------------------------------------------------------------------
// Kernel 2: flash attention. One wave per 16-query tile per (b,h).
// grid = B*H*(S/16)/4 blocks of 128 threads.
// Pipeline: score WMMAs consume K-fragments prefetched last iteration;
// next chunk's K-fragments and this chunk's V-fragments are issued before
// the softmax VALU section so their latency hides under exp/shuffles.
// ---------------------------------------------------------------------------
__global__ __launch_bounds__(128) void flash_attn_kernel(
    const _Float16* __restrict__ Qh, const _Float16* __restrict__ Kh,
    const _Float16* __restrict__ Vt, const int* __restrict__ mask,
    _Float16* __restrict__ Oh) {

    __shared__ _Float16 ptile[4][16 * 32];

    const int lane = threadIdx.x & 31;
    const int wave = threadIdx.x >> 5;
    const int wid  = blockIdx.x * 4 + wave;
    const int qt   = wid & (Sv / 16 - 1);   // S/16 = 128 tiles
    const int bh   = wid >> 7;
    const int b    = bh / Hv;
    const int h    = bh % Hv;
    const int q0   = qt * 16;

    const _Float16* Qbase = Qh + (size_t)bh * Sv * HDv;
    const _Float16* Kbase = Kh + (size_t)bh * Sv * HDv;
    const _Float16* Vbase = Vt + (size_t)bh * HDv * Sv;
    const int* mrow = mask + b * Sv;

    // Q A-fragments for the whole tile (HD=64 -> two K=32 fragments)
    const v16h qa0 = load_a_frag(Qbase + (size_t)q0 * HDv + 0,  HDv, lane);
    const v16h qa1 = load_a_frag(Qbase + (size_t)q0 * HDv + 32, HDv, lane);

    v8f o[4] = {v8f{}, v8f{}, v8f{}, v8f{}};
    float m_r[8], l_r[8];
#pragma unroll
    for (int r = 0; r < 8; ++r) { m_r[r] = -1e30f; l_r[r] = 0.0f; }

    const int half = lane >> 4;
    const int cl   = lane & 15;
    _Float16* pt = ptile[wave];

    // preload chunk 0 K-fragments
    v16h kb0 = load_b_frag(Kbase + 0,                        HDv, lane);
    v16h kb1 = load_b_frag(Kbase + 32,                       HDv, lane);
    v16h kb2 = load_b_frag(Kbase + (size_t)16 * HDv + 0,     HDv, lane);
    v16h kb3 = load_b_frag(Kbase + (size_t)16 * HDv + 32,    HDv, lane);

    for (int kc = 0; kc < Sv; kc += 32) {
        // ---- scores from prefetched K-fragments ----
        v8f s0 = v8f{}, s1 = v8f{};
        s0 = wmma_f16(qa0, kb0, s0);
        s0 = wmma_f16(qa1, kb1, s0);
        s1 = wmma_f16(qa0, kb2, s1);
        s1 = wmma_f16(qa1, kb3, s1);

        // ---- issue next chunk's K-fragments + this chunk's V-fragments ----
        const int kn = (kc + 32 < Sv) ? (kc + 32) : kc;
        v16h nkb0 = load_b_frag(Kbase + (size_t)kn * HDv + 0,         HDv, lane);
        v16h nkb1 = load_b_frag(Kbase + (size_t)kn * HDv + 32,        HDv, lane);
        v16h nkb2 = load_b_frag(Kbase + (size_t)(kn + 16) * HDv + 0,  HDv, lane);
        v16h nkb3 = load_b_frag(Kbase + (size_t)(kn + 16) * HDv + 32, HDv, lane);
        v16h vb0 = load_b_frag(Vbase + (size_t)0  * Sv + kc, Sv, lane);
        v16h vb1 = load_b_frag(Vbase + (size_t)16 * Sv + kc, Sv, lane);
        v16h vb2 = load_b_frag(Vbase + (size_t)32 * Sv + kc, Sv, lane);
        v16h vb3 = load_b_frag(Vbase + (size_t)48 * Sv + kc, Sv, lane);

        // scale by 1/sqrt(64) and apply key mask (lane's column = one key)
        const bool mz0 = (mrow[kc + cl] == 0);
        const bool mz1 = (mrow[kc + 16 + cl] == 0);
#pragma unroll
        for (int r = 0; r < 8; ++r) {
            float v0 = s0[r] * 0.125f; if (mz0) v0 = -1e9f;
            float v1 = s1[r] * 0.125f; if (mz1) v1 = -1e9f;
            s0[r] = v0; s1[r] = v1;
        }

        // ---- online softmax (row = M lives across the 16-lane half) ----
        float p0[8], p1[8], alpha[8];
#pragma unroll
        for (int r = 0; r < 8; ++r) {
            float rm = fmaxf(s0[r], s1[r]);
            rm = fmaxf(rm, __shfl_xor(rm, 1, 32));
            rm = fmaxf(rm, __shfl_xor(rm, 2, 32));
            rm = fmaxf(rm, __shfl_xor(rm, 4, 32));
            rm = fmaxf(rm, __shfl_xor(rm, 8, 32));
            const float mn = fmaxf(m_r[r], rm);
            alpha[r] = __expf(m_r[r] - mn);
            m_r[r]   = mn;
            p0[r] = __expf(s0[r] - mn);
            p1[r] = __expf(s1[r] - mn);
            float rs = p0[r] + p1[r];
            rs += __shfl_xor(rs, 1, 32);
            rs += __shfl_xor(rs, 2, 32);
            rs += __shfl_xor(rs, 4, 32);
            rs += __shfl_xor(rs, 8, 32);
            l_r[r] = l_r[r] * alpha[r] + rs;
        }
#pragma unroll
        for (int r = 0; r < 8; ++r) {
            o[0][r] *= alpha[r]; o[1][r] *= alpha[r];
            o[2][r] *= alpha[r]; o[3][r] *= alpha[r];
        }

        // ---- stage P (16x32) through LDS to build the A-fragment ----
#pragma unroll
        for (int r = 0; r < 8; ++r) {
            const int row = r + 8 * half;
            pt[row * 32 + cl]      = (_Float16)p0[r];
            pt[row * 32 + 16 + cl] = (_Float16)p1[r];
        }
        // LDS ops from the same wave execute in order (DScnt): no barrier needed.
        const v16h pa = load_a_frag(pt, 32, lane);

        // ---- O += P @ V  (V stored transposed: rows = d, contiguous keys) ----
        o[0] = wmma_f16(pa, vb0, o[0]);
        o[1] = wmma_f16(pa, vb1, o[1]);
        o[2] = wmma_f16(pa, vb2, o[2]);
        o[3] = wmma_f16(pa, vb3, o[3]);

        kb0 = nkb0; kb1 = nkb1; kb2 = nkb2; kb3 = nkb3;
    }

    // ---- normalize and write O (B,S,D) as f16 ----
#pragma unroll
    for (int r = 0; r < 8; ++r) {
        const float inv = 1.0f / l_r[r];
        o[0][r] *= inv; o[1][r] *= inv; o[2][r] *= inv; o[3][r] *= inv;
    }
#pragma unroll
    for (int f = 0; f < 4; ++f) {
#pragma unroll
        for (int r = 0; r < 8; ++r) {
            const int s   = q0 + r + 8 * half;
            const int col = h * HDv + f * 16 + cl;
            Oh[((size_t)b * Sv + s) * Dv + col] = (_Float16)o[f][r];
        }
    }
}

// ---------------------------------------------------------------------------
// Kernel 3: output projection, fp32 result. grid = (BS/32, 3), block = 128.
// Same 32x64 software-pipelined tiling as the QKV projection.
// ---------------------------------------------------------------------------
__global__ __launch_bounds__(128) void out_proj_kernel(
    const _Float16* __restrict__ Oh, const _Float16* __restrict__ woh,
    const float* __restrict__ bo, float* __restrict__ out) {

    const int lane = threadIdx.x & 31;
    const int wave = threadIdx.x >> 5;
    const int m0 = blockIdx.x * 32;
    const int n0 = blockIdx.y * 256 + wave * 64;

    const _Float16* abase = Oh  + (size_t)m0 * Dv;
    const _Float16* bbase = woh + (size_t)n0 * Dv;

    v8f acc[8] = {v8f{}, v8f{}, v8f{}, v8f{}, v8f{}, v8f{}, v8f{}, v8f{}};

    v16h a0 = load_a_frag(abase, Dv, lane);
    v16h a1 = load_a_frag(abase + (size_t)16 * Dv, Dv, lane);
    v16h b0 = load_b_frag(bbase + (size_t)0  * Dv, Dv, lane);
    v16h b1 = load_b_frag(bbase + (size_t)16 * Dv, Dv, lane);
    v16h b2 = load_b_frag(bbase + (size_t)32 * Dv, Dv, lane);
    v16h b3 = load_b_frag(bbase + (size_t)48 * Dv, Dv, lane);

    for (int k0 = 0; k0 < Dv; k0 += 32) {
        const int kn = (k0 + 32 < Dv) ? (k0 + 32) : k0;
        v16h na0 = load_a_frag(abase + kn, Dv, lane);
        v16h na1 = load_a_frag(abase + (size_t)16 * Dv + kn, Dv, lane);
        v16h nb0 = load_b_frag(bbase + (size_t)0  * Dv + kn, Dv, lane);
        v16h nb1 = load_b_frag(bbase + (size_t)16 * Dv + kn, Dv, lane);
        v16h nb2 = load_b_frag(bbase + (size_t)32 * Dv + kn, Dv, lane);
        v16h nb3 = load_b_frag(bbase + (size_t)48 * Dv + kn, Dv, lane);

        acc[0] = wmma_f16(a0, b0, acc[0]);
        acc[1] = wmma_f16(a0, b1, acc[1]);
        acc[2] = wmma_f16(a0, b2, acc[2]);
        acc[3] = wmma_f16(a0, b3, acc[3]);
        acc[4] = wmma_f16(a1, b0, acc[4]);
        acc[5] = wmma_f16(a1, b1, acc[5]);
        acc[6] = wmma_f16(a1, b2, acc[6]);
        acc[7] = wmma_f16(a1, b3, acc[7]);

        a0 = na0; a1 = na1; b0 = nb0; b1 = nb1; b2 = nb2; b3 = nb3;
    }

    const int half = lane >> 4;
    const int cl   = lane & 15;
#pragma unroll
    for (int f = 0; f < 4; ++f) {
        const int n = n0 + f * 16 + cl;
        const float bval = bo[n];
#pragma unroll
        for (int r = 0; r < 8; ++r) {
            const int m = m0 + r + 8 * half;
            out[(size_t)m * Dv + n]        = acc[f][r]     + bval;
            out[(size_t)(m + 16) * Dv + n] = acc[4 + f][r] + bval;
        }
    }
}

// ---------------------------------------------------------------------------
// Host-side launch
// ---------------------------------------------------------------------------
extern "C" void kernel_launch(void* const* d_in, const int* in_sizes, int n_in,
                              void* d_out, int out_size, void* d_ws, size_t ws_size,
                              hipStream_t stream) {
    const float* x  = (const float*)d_in[0];
    const int*   mk = (const int*)  d_in[1];
    const float* wq = (const float*)d_in[2];
    const float* bq = (const float*)d_in[3];
    const float* wk = (const float*)d_in[4];
    const float* bk = (const float*)d_in[5];
    const float* wv = (const float*)d_in[6];
    const float* bv = (const float*)d_in[7];
    const float* wo = (const float*)d_in[8];
    const float* bo = (const float*)d_in[9];
    float* out = (float*)d_out;

    const size_t NX = (size_t)Bv * Sv * Dv;   // 6,291,456
    const size_t NW = (size_t)Dv * Dv;        //   589,824

    _Float16* p   = (_Float16*)d_ws;
    _Float16* xh  = p; p += NX;
    _Float16* wqh = p; p += NW;
    _Float16* wkh = p; p += NW;
    _Float16* wvh = p; p += NW;
    _Float16* woh = p; p += NW;
    _Float16* Qh  = p; p += NX;   // (B,H,S,HD)
    _Float16* Kh  = p; p += NX;   // (B,H,S,HD)
    _Float16* Vt  = p; p += NX;   // (B,H,HD,S)
    _Float16* Oh  = p; p += NX;   // (B,S,D)

    // 0) fp32 -> f16
    {
        const size_t total = NX + 4 * NW;
        const int threads = 256;
        const int blocks = (int)((total + threads - 1) / threads);
        convert_f16_kernel<<<blocks, threads, 0, stream>>>(
            x, wq, wk, wv, wo, xh, wqh, wkh, wvh, woh);
    }
    // 1) QKV projections
    {
        dim3 grid(Bv * Sv / 32, 3, 3);   // (256, 3, 3)
        qkv_proj_kernel<<<grid, 128, 0, stream>>>(
            xh, wqh, wkh, wvh, bq, bk, bv, Qh, Kh, Vt);
    }
    // 2) flash attention
    {
        const int nwaves = Bv * Hv * (Sv / 16);   // 6144
        flash_attn_kernel<<<nwaves / 4, 128, 0, stream>>>(Qh, Kh, Vt, mk, Oh);
    }
    // 3) output projection
    {
        dim3 grid(Bv * Sv / 32, 3, 1);   // (256, 3)
        out_proj_kernel<<<grid, 128, 0, stream>>>(Oh, woh, bo, out);
    }
}
